// SelfAttentiom_4131758539222
// MI455X (gfx1250) — matrix operand order repaired and checked by reading, not compiled, not run on hardware
//
#include <hip/hip_runtime.h>

// ---------------------------------------------------------------------------
// Self-attention for MI455X (gfx1250): bf16 WMMA everywhere.
// B=4, S=2048, D=1024, H=16, dh=64.
// ---------------------------------------------------------------------------

typedef __attribute__((ext_vector_type(16))) __bf16 v16bf;
typedef __attribute__((ext_vector_type(8)))  float  v8f;
typedef __attribute__((__vector_size__(16))) int    v4i;
typedef __attribute__((__vector_size__(8)))  int    v2i;

#define NB 4
#define NS 2048
#define ND 1024
#define NH 16
#define NDH 64
#define NBS 8192          // B*S
#define NE3 3072          // 3*D

#define GLOBAL_AS __attribute__((address_space(1)))
#define LDS_AS    __attribute__((address_space(3)))

#if defined(__has_builtin)
#if __has_builtin(__builtin_amdgcn_cvt_pk_bf16_f32)
#define USE_CVT_PK_BF16 1
#endif
#if __has_builtin(__builtin_amdgcn_global_load_async_to_lds_b128) && \
    __has_builtin(__builtin_amdgcn_global_load_async_to_lds_b64) &&  \
    __has_builtin(__builtin_amdgcn_s_wait_asynccnt)
#define HAVE_ASYNC_LDS 1
#endif
#endif

union Frag16 {
    v16bf v;
    uint4 q[2];
    unsigned short u[16];
};

// f32 -> bf16 RNE via native fptrunc (backend picks the HW cvt when available)
static __device__ __forceinline__ unsigned short f2bf(float f) {
    union { __bf16 h; unsigned short u; } c;
    c.h = (__bf16)f;
    return c.u;
}

static __device__ __forceinline__ unsigned int pack2bf(float a, float b) {
#ifdef USE_CVT_PK_BF16
    typedef __attribute__((ext_vector_type(2))) __bf16 v2bf;
    union { v2bf v; unsigned int u; } c;
    c.v = __builtin_amdgcn_cvt_pk_bf16_f32(a, b);
    return c.u;
#else
    return (unsigned int)f2bf(a) | ((unsigned int)f2bf(b) << 16);
#endif
}

// A-fragment (16x32 bf16): lane row = lane&15, kh = lane>>4.
// elements 0..15 -> K = kh*16 + 0..15
static __device__ __forceinline__ v16bf load_frag_a(const unsigned short* rowp, int kh) {
    Frag16 f;
    f.q[0] = *reinterpret_cast<const uint4*>(rowp + kh * 16);
    f.q[1] = *reinterpret_cast<const uint4*>(rowp + kh * 16 + 8);
    return f.v;
}

static __device__ __forceinline__ v8f wmma_bf16(v16bf a, v16bf b, v8f c) {
    return __builtin_amdgcn_wmma_f32_16x16x32_bf16(
        false, a, false, b, (short)0, c, false, false);
}

// Stage 32 bytes (16 bf16) global -> LDS (async DMA when available).
static __device__ __forceinline__ void stage32B(const unsigned short* g,
                                                unsigned short* l) {
#ifdef HAVE_ASYNC_LDS
    __builtin_amdgcn_global_load_async_to_lds_b128(
        (GLOBAL_AS v4i*)g, (LDS_AS v4i*)l, 0, 0);
    __builtin_amdgcn_global_load_async_to_lds_b128(
        (GLOBAL_AS v4i*)(g + 8), (LDS_AS v4i*)(l + 8), 0, 0);
#else
    const uint4* p = reinterpret_cast<const uint4*>(g);
    uint4* d = reinterpret_cast<uint4*>(l);
    d[0] = p[0];
    d[1] = p[1];
#endif
}
static __device__ __forceinline__ void stage8B(const unsigned short* g,
                                               unsigned short* l) {
#ifdef HAVE_ASYNC_LDS
    __builtin_amdgcn_global_load_async_to_lds_b64(
        (GLOBAL_AS v2i*)g, (LDS_AS v2i*)l, 0, 0);
#else
    *reinterpret_cast<uint2*>(l) = *reinterpret_cast<const uint2*>(g);
#endif
}
static __device__ __forceinline__ void stage_wait() {
#ifdef HAVE_ASYNC_LDS
    __builtin_amdgcn_s_wait_asynccnt(0);
#endif
}

// ---------------------------------------------------------------------------
// Kernel 0: fp32 -> bf16 conversion (2 elems/thread, packed stores)
// ---------------------------------------------------------------------------
__global__ __launch_bounds__(256) void cvt_f32_bf16_kernel(
    const float* __restrict__ in, unsigned int* __restrict__ out, int npairs) {
    int i = blockIdx.x * 256 + threadIdx.x;
    if (i < npairs) {
        float2 f = reinterpret_cast<const float2*>(in)[i];
        out[i] = pack2bf(f.x, f.y);
    }
}

// ---------------------------------------------------------------------------
// Unified bf16 GEMM: C[N=8192, E] = A(bf16)[N,1024] @ W(bf16)[E,1024]^T
// 256 threads = 8 waves; 128x128 C tile; BK=32; double-buffered LDS with
// (optionally async) global->LDS staging. Wave grid 2x4, 4x2 WMMA tiles/wave.
// ---------------------------------------------------------------------------
template <int EOUT, bool OUT_BF16>
__global__ __launch_bounds__(256) void gemm_bf16_kernel(
    const unsigned short* __restrict__ A, const unsigned short* __restrict__ W,
    void* __restrict__ Cv) {
    __shared__ unsigned short As[2][128 * 32];
    __shared__ unsigned short Bs[2][128 * 32];

    const int tid  = threadIdx.x;
    const int wid  = tid >> 5;
    const int lane = tid & 31;
    const int widM = wid >> 2;       // 0..1
    const int widN = wid & 3;        // 0..3
    const int lrow = lane & 15;
    const int kh   = lane >> 4;
    const int bm   = blockIdx.x;
    const int bn   = blockIdx.y;

    v8f acc[4][2];
    const v8f vzero = {0.f,0.f,0.f,0.f,0.f,0.f,0.f,0.f};
#pragma unroll
    for (int mt = 0; mt < 4; ++mt)
#pragma unroll
        for (int nt = 0; nt < 2; ++nt) acc[mt][nt] = vzero;

    const int rowL = tid >> 1;              // 0..127
    const int kseg = (tid & 1) * 16;        // 0 or 16
    const unsigned short* arow = A + (size_t)(bm * 128 + rowL) * ND;
    const unsigned short* wrow = W + (size_t)(bn * 128 + rowL) * ND;

    // ---- prologue: stage k-slice 0 into buffer 0 ----
    stage32B(arow + kseg, &As[0][rowL * 32 + kseg]);
    stage32B(wrow + kseg, &Bs[0][rowL * 32 + kseg]);
    stage_wait();
    __syncthreads();

    int cur = 0;
    for (int kk = 0; kk < ND; kk += 32) {
        const bool hasNext = (kk + 32) < ND;
        const int nxt = cur ^ 1;
#ifdef HAVE_ASYNC_LDS
        if (hasNext) {   // issue DMA before compute so it overlaps the WMMAs
            stage32B(arow + kk + 32 + kseg, &As[nxt][rowL * 32 + kseg]);
            stage32B(wrow + kk + 32 + kseg, &Bs[nxt][rowL * 32 + kseg]);
        }
#else
        uint4 fa[2], fb[2];
        if (hasNext) {
            const uint4* pa = reinterpret_cast<const uint4*>(arow + kk + 32 + kseg);
            fa[0] = pa[0];  fa[1] = pa[1];
            const uint4* pb = reinterpret_cast<const uint4*>(wrow + kk + 32 + kseg);
            fb[0] = pb[0];  fb[1] = pb[1];
        }
#endif

        // ---- compute from buffer `cur` ----
        v16bf bfr[2];
#pragma unroll
        for (int nt = 0; nt < 2; ++nt)
            bfr[nt] = *reinterpret_cast<const v16bf*>(
                &Bs[cur][(widN * 32 + nt * 16 + lrow) * 32 + kh * 16]);
#pragma unroll
        for (int mt = 0; mt < 4; ++mt) {
            v16bf afr = load_frag_a(&As[cur][(widM * 64 + mt * 16 + lrow) * 32], kh);
#pragma unroll
            for (int nt = 0; nt < 2; ++nt)
                acc[mt][nt] = wmma_bf16(afr, bfr[nt], acc[mt][nt]);
        }

#ifndef HAVE_ASYNC_LDS
        if (hasNext) {
            uint4* asd = reinterpret_cast<uint4*>(&As[nxt][rowL * 32 + kseg]);
            asd[0] = fa[0];  asd[1] = fa[1];
            uint4* bsd = reinterpret_cast<uint4*>(&Bs[nxt][rowL * 32 + kseg]);
            bsd[0] = fb[0];  bsd[1] = fb[1];
        }
#endif
        if (hasNext) stage_wait();
        __syncthreads();
        cur = nxt;
    }

    // ---- store ----
#pragma unroll
    for (int mt = 0; mt < 4; ++mt)
#pragma unroll
        for (int nt = 0; nt < 2; ++nt)
#pragma unroll
            for (int r = 0; r < 8; ++r) {
                int grow = bm * 128 + widM * 64 + mt * 16 + r + 8 * kh;
                int gcol = bn * 128 + widN * 32 + nt * 16 + lrow;
                if (OUT_BF16)
                    ((unsigned short*)Cv)[(size_t)grow * EOUT + gcol] =
                        f2bf(acc[mt][nt][r]);
                else
                    ((float*)Cv)[(size_t)grow * EOUT + gcol] = acc[mt][nt][r];
            }
}

// ---------------------------------------------------------------------------
// Flash attention. 1 block = (b,h,qtile of 64 rows), 128 thr = 4 waves.
// Wave w owns q rows [qbase + w*16, +16). Online softmax over 64-key tiles.
// V is staged TRANSPOSED so PV B-fragments are contiguous b128 loads.
// ---------------------------------------------------------------------------
#define VT_STRIDE 80   // padded key-stride of transposed V (multiple of 16)

__global__ __launch_bounds__(128) void flash_attn_kernel(
    const unsigned short* __restrict__ qkv, unsigned short* __restrict__ aout,
    const int* __restrict__ cmask) {
    __shared__ unsigned short Qs[64 * 64];
    __shared__ unsigned short Ks[64 * 64];
    __shared__ unsigned short Vt[64 * VT_STRIDE];  // [dh][key]
    __shared__ unsigned short Ps[4 * 16 * 64];     // per-wave P stripe

    const int tid  = threadIdx.x;
    const int wid  = tid >> 5;
    const int lane = tid & 31;
    const int lrow = lane & 15;
    const int kh   = lane >> 4;

    const int qt = blockIdx.x & 31;          // S/64 = 32 q-tiles
    const int bh = blockIdx.x >> 5;
    const int b  = bh >> 4;
    const int h  = bh & 15;
    const int qbase = qt * 64;
    const int causal = *cmask;

    // ---- load Q tile (64x64 bf16), async when available ----
#pragma unroll
    for (int i = 0; i < 8; ++i) {
        int g   = i * 128 + tid;
        int row = g >> 4;
        int dc  = (g & 15) * 4;
        stage8B(qkv + (size_t)(b * NS + qbase + row) * NE3 + h * NDH + dc,
                &Qs[row * 64 + dc]);
    }
    stage_wait();
    __syncthreads();

    v16bf qfrag[2];
    {
        const unsigned short* qr = &Qs[(wid * 16 + lrow) * 64];
#pragma unroll
        for (int kc = 0; kc < 2; ++kc) qfrag[kc] = load_frag_a(qr + kc * 32, kh);
    }

    const v8f vzero = {0.f,0.f,0.f,0.f,0.f,0.f,0.f,0.f};
    v8f Oacc[4];
#pragma unroll
    for (int nt = 0; nt < 4; ++nt) Oacc[nt] = vzero;
    float mrow[8], lrowsum[8];
#pragma unroll
    for (int r = 0; r < 8; ++r) { mrow[r] = -1e30f; lrowsum[r] = 0.f; }

    const int ktmax = causal ? qt : 31;
    for (int kt = 0; kt <= ktmax; ++kt) {
        // ---- stage K (row-major, async) and V (transposed, via regs) ----
#pragma unroll
        for (int i = 0; i < 8; ++i) {
            int g   = i * 128 + tid;
            int row = g >> 4;          // key index
            int dc  = (g & 15) * 4;    // dh column start
            size_t base = (size_t)(b * NS + kt * 64 + row) * NE3 + h * NDH + dc;
            stage8B(qkv + base + ND, &Ks[row * 64 + dc]);
            uint2 v = *reinterpret_cast<const uint2*>(qkv + base + 2 * ND);
            Vt[(dc + 0) * VT_STRIDE + row] = (unsigned short)(v.x & 0xffffu);
            Vt[(dc + 1) * VT_STRIDE + row] = (unsigned short)(v.x >> 16);
            Vt[(dc + 2) * VT_STRIDE + row] = (unsigned short)(v.y & 0xffffu);
            Vt[(dc + 3) * VT_STRIDE + row] = (unsigned short)(v.y >> 16);
        }
        stage_wait();
        __syncthreads();

        // ---- S = Q K^T : wave stripe 16x64 ----
        v8f sacc[4];
#pragma unroll
        for (int nt = 0; nt < 4; ++nt) sacc[nt] = vzero;
#pragma unroll
        for (int nt = 0; nt < 4; ++nt)
#pragma unroll
            for (int kc = 0; kc < 2; ++kc) {
                v16bf bfr = *reinterpret_cast<const v16bf*>(
                    &Ks[(nt * 16 + lrow) * 64 + kc * 32 + kh * 16]);
                sacc[nt] = wmma_bf16(qfrag[kc], bfr, sacc[nt]);
            }

        // ---- online softmax per row (rows live in 16-lane halves) ----
#pragma unroll
        for (int r = 0; r < 8; ++r) {
            int rowLocal = r + 8 * kh;
            int qrow = qbase + wid * 16 + rowLocal;
            float sv[4];
            float rmax = -1e30f;
#pragma unroll
            for (int nt = 0; nt < 4; ++nt) {
                int key = kt * 64 + nt * 16 + lrow;
                float s = sacc[nt][r] * 0.125f;          // 1/sqrt(64)
                if (causal && key > qrow) s = -1e30f;
                sv[nt] = s;
                rmax = fmaxf(rmax, s);
            }
#pragma unroll
            for (int m = 1; m < 16; m <<= 1)
                rmax = fmaxf(rmax, __shfl_xor(rmax, m, 32));
            float mnew = fmaxf(mrow[r], rmax);
            float corr = __expf(mrow[r] - mnew);
            float psum = 0.f;
#pragma unroll
            for (int nt = 0; nt < 4; ++nt) {
                float p = __expf(sv[nt] - mnew);
                psum += p;
                Ps[wid * 1024 + rowLocal * 64 + nt * 16 + lrow] = f2bf(p);
            }
#pragma unroll
            for (int m = 1; m < 16; m <<= 1) psum += __shfl_xor(psum, m, 32);
            lrowsum[r] = lrowsum[r] * corr + psum;
            mrow[r] = mnew;
#pragma unroll
            for (int nt = 0; nt < 4; ++nt) Oacc[nt][r] = Oacc[nt][r] * corr;
        }

        // ---- O += P V ----
        v16bf pfrag[2];
        {
            const unsigned short* pr = &Ps[wid * 1024 + lrow * 64];
#pragma unroll
            for (int kc = 0; kc < 2; ++kc) pfrag[kc] = load_frag_a(pr + kc * 32, kh);
        }
#pragma unroll
        for (int nt = 0; nt < 4; ++nt)
#pragma unroll
            for (int kc = 0; kc < 2; ++kc) {
                v16bf vfr = *reinterpret_cast<const v16bf*>(
                    &Vt[(nt * 16 + lrow) * VT_STRIDE + kc * 32 + kh * 16]);
                Oacc[nt] = wmma_bf16(pfrag[kc], vfr, Oacc[nt]);
            }
        __syncthreads();
    }

    // ---- normalize + store bf16 attn-out [BS, D] ----
#pragma unroll
    for (int r = 0; r < 8; ++r) {
        int rowLocal = r + 8 * kh;
        float rinv = 1.0f / lrowsum[r];
        int token = b * NS + qbase + wid * 16 + rowLocal;
#pragma unroll
        for (int nt = 0; nt < 4; ++nt) {
            int col = h * NDH + nt * 16 + lrow;
            aout[(size_t)token * ND + col] = f2bf(Oacc[nt][r] * rinv);
        }
    }
}

// ---------------------------------------------------------------------------
// Host launcher
// ---------------------------------------------------------------------------
extern "C" void kernel_launch(void* const* d_in, const int* in_sizes, int n_in,
                              void* d_out, int out_size, void* d_ws, size_t ws_size,
                              hipStream_t stream) {
    const float* x     = (const float*)d_in[0];   // [4,2048,1024]
    const float* w_in  = (const float*)d_in[1];   // [3072,1024]
    const float* w_out = (const float*)d_in[2];   // [1024,1024]
    const int*   cm    = (const int*)d_in[3];     // causal flag

    // workspace layout (bytes)
    char* ws = (char*)d_ws;
    unsigned short* win_bf  = (unsigned short*)(ws);                 //  6 MB
    unsigned short* wout_bf = (unsigned short*)(ws + 6291456);       //  2 MB
    unsigned short* x_bf    = (unsigned short*)(ws + 8388608);       // 16 MB
    unsigned short* qkv_bf  = (unsigned short*)(ws + 25165824);      // 48 MB
    unsigned short* attn_bf = (unsigned short*)(ws + 75497472);      // 16 MB

    // 0) conversions fp32 -> bf16 (weights + activations)
    cvt_f32_bf16_kernel<<<(NE3 * ND / 2 + 255) / 256, 256, 0, stream>>>(
        w_in, (unsigned int*)win_bf, NE3 * ND / 2);
    cvt_f32_bf16_kernel<<<(ND * ND / 2 + 255) / 256, 256, 0, stream>>>(
        w_out, (unsigned int*)wout_bf, ND * ND / 2);
    cvt_f32_bf16_kernel<<<(NBS * ND / 2 + 255) / 256, 256, 0, stream>>>(
        x, (unsigned int*)x_bf, NBS * ND / 2);

    // 1) fused QKV projection (bf16 out)
    gemm_bf16_kernel<NE3, true><<<dim3(NBS / 128, NE3 / 128), 256, 0, stream>>>(
        x_bf, win_bf, qkv_bf);

    // 2) flash attention: B*H*(S/64) = 2048 blocks of 128 threads
    flash_attn_kernel<<<NB * NH * (NS / 64), 128, 0, stream>>>(qkv_bf, attn_bf, cm);

    // 3) output projection (fp32 out)
    gemm_bf16_kernel<ND, false><<<dim3(NBS / 128, ND / 128), 256, 0, stream>>>(
        attn_bf, wout_bf, (float*)d_out);
}
